// NALU_52243982189021
// MI455X (gfx1250) — compile-verified, matrix operand order checked by reading, not run on hardware
//
#include <hip/hip_runtime.h>
#include <cstdint>

typedef __attribute__((ext_vector_type(16))) __bf16 v16bf;
typedef __attribute__((ext_vector_type(8)))  float  v8f;
typedef __attribute__((ext_vector_type(4)))  uint32_t u32x4;
typedef __attribute__((ext_vector_type(8)))  uint32_t u32x8;

#define B_DIM 8192
#define K_DIM 2048
#define N_DIM 2048
#define LOG2E 1.442695041f
#define LN2   0.6931471806f
#define LDSROW 40   // 32 bf16 + 8 pad elements = 80 B row stride (conflict-free)

// ---------- helpers ----------

__device__ __forceinline__ uint16_t f32_to_bf16(float f) {
  uint32_t u = __float_as_uint(f);
  uint32_t r = u + 0x7FFFu + ((u >> 16) & 1u);   // round-to-nearest-even
  return (uint16_t)(r >> 16);
}

__device__ __forceinline__ float fast_exp(float x) {
  return __builtin_amdgcn_exp2f(x * LOG2E);
}
__device__ __forceinline__ float fast_sigmoid(float x) {
  return __builtin_amdgcn_rcpf(1.0f + __builtin_amdgcn_exp2f(x * -LOG2E));
}

// Load one 16x32 bf16 WMMA fragment (per-lane: 16 bf16 = two b128 loads).
__device__ __forceinline__ v16bf load_frag(const uint16_t* p) {
  union { uint4 q[2]; v16bf v; } u;
  u.q[0] = *(const uint4*)(p);        // K 0..7   (lanes 0-15) / K 8..15  (lanes 16-31)
  u.q[1] = *(const uint4*)(p + 16);   // K 16..23 (lanes 0-15) / K 24..31 (lanes 16-31)
  return u.v;
}

// TDM descriptor group 0: count=1, lds_addr, 57-bit global tile addr, type=2.
__device__ __forceinline__ u32x4 tdm_g0(uint32_t lds_off, uint64_t ga) {
  u32x4 g;
  g[0] = 1u;
  g[1] = lds_off;
  g[2] = (uint32_t)ga;
  g[3] = (uint32_t)((ga >> 32) & 0x01FFFFFFu) | 0x80000000u;  // type=2 ("image")
  return g;
}

// ---------- prep 1: x -> bf16(x), bf16(log(|x|+eps)) ----------

__global__ void nalu_prep_x(const float* __restrict__ x,
                            uint16_t* __restrict__ xb,
                            uint16_t* __restrict__ lb) {
  size_t i = ((size_t)blockIdx.x * blockDim.x + threadIdx.x) * 4;
  float4 v = *(const float4*)(x + i);
  uint32_t x0 = (uint32_t)f32_to_bf16(v.x) | ((uint32_t)f32_to_bf16(v.y) << 16);
  uint32_t x1 = (uint32_t)f32_to_bf16(v.z) | ((uint32_t)f32_to_bf16(v.w) << 16);
  float l0 = LN2 * __builtin_amdgcn_logf(fabsf(v.x) + 1e-7f);
  float l1 = LN2 * __builtin_amdgcn_logf(fabsf(v.y) + 1e-7f);
  float l2 = LN2 * __builtin_amdgcn_logf(fabsf(v.z) + 1e-7f);
  float l3 = LN2 * __builtin_amdgcn_logf(fabsf(v.w) + 1e-7f);
  uint32_t y0 = (uint32_t)f32_to_bf16(l0) | ((uint32_t)f32_to_bf16(l1) << 16);
  uint32_t y1 = (uint32_t)f32_to_bf16(l2) | ((uint32_t)f32_to_bf16(l3) << 16);
  *(uint2*)(xb + i) = make_uint2(x0, x1);
  *(uint2*)(lb + i) = make_uint2(y0, y1);
}

// ---------- prep 2: W = tanh(W_hat)*sigmoid(M_hat); transpose W, gate to [OUT][IN] bf16 ----------

__global__ void nalu_prep_w(const float* __restrict__ Wh,
                            const float* __restrict__ Mh,
                            const float* __restrict__ G,
                            uint16_t* __restrict__ Wt,
                            uint16_t* __restrict__ Gt) {
  __shared__ uint16_t sW[32][33];
  __shared__ uint16_t sG[32][33];
  const int tx = threadIdx.x;            // 0..31
  const int ty = threadIdx.y;            // 0..7
  const int c0 = blockIdx.x * 32;        // OUT tile base
  const int r0 = blockIdx.y * 32;        // IN tile base
#pragma unroll
  for (int i = 0; i < 4; ++i) {
    int r = r0 + ty + i * 8;
    int c = c0 + tx;
    size_t idx = (size_t)r * N_DIM + c;  // coalesced along tx
    float w = tanhf(Wh[idx]) * fast_sigmoid(Mh[idx]);
    sW[tx][ty + i * 8] = f32_to_bf16(w);   // store transposed into LDS
    sG[tx][ty + i * 8] = f32_to_bf16(G[idx]);
  }
  __syncthreads();
#pragma unroll
  for (int i = 0; i < 4; ++i) {
    int oc = c0 + ty + i * 8;            // output row in Wt = original column
    size_t idx = (size_t)oc * K_DIM + (r0 + tx);  // coalesced along tx
    Wt[idx] = sW[ty + i * 8][tx];
    Gt[idx] = sG[ty + i * 8][tx];
  }
}

// ---------- fused triple-GEMM + NALU epilogue ----------
// out = g*a + (1-g)*m ;  a = x@W, m = exp(logx@W), g = sigmoid(x@gate)
// A operands (x, logx): GLOBAL_LOAD_ASYNC_TO_LDS_B128 pipeline (ASYNCcnt).
// B operands (W^T, gate^T): Tensor Data Mover tensor_load_to_lds (TENSORcnt),
// with descriptor-side LDS padding for bank-conflict-free ds_load_b128.
// Pipeline is hand-unrolled x2 so buffer indices are compile-time constants.

__global__ __launch_bounds__(256) void nalu_gemm(
    const uint16_t* __restrict__ xb,   // [B][K] bf16
    const uint16_t* __restrict__ lb,   // [B][K] bf16 log path
    const uint16_t* __restrict__ wt,   // [N][K] bf16 W^T
    const uint16_t* __restrict__ gt,   // [N][K] bf16 gate^T
    float* __restrict__ out) {         // [B][N] f32
  __shared__ uint16_t sX[2][128 * LDSROW];   // 10 KB / buf
  __shared__ uint16_t sL[2][128 * LDSROW];   // 10 KB / buf
  __shared__ uint16_t sW[2][64 * LDSROW];    //  5 KB / buf
  __shared__ uint16_t sG[2][64 * LDSROW];    //  5 KB / buf

  const int tid  = threadIdx.x;
  const int lane = tid & 31;
  const int wave = tid >> 5;           // 0..7
  const int lr   = lane & 15;
  const int lh   = lane >> 4;          // selects K 8..15 / 24..31 slice
  const int wm   = wave & 3;           // 4 waves along M
  const int wn   = wave >> 2;          // 2 waves along N
  const int nb   = blockIdx.x * 64;
  const int mb   = blockIdx.y * 128;
  const int m0   = mb + wm * 32;

  // ---- A staging geometry: 128x32 bf16 tile = 512 x 16B chunks; 256 threads
  //      -> 2 chunks per thread per matrix (4 async ops / thread / slab).
  const uint16_t* gx[2];
  const uint16_t* gl[2];
  uint32_t ldx0[2], ldl0[2];           // buf0 padded LDS byte offsets
#pragma unroll
  for (int h = 0; h < 2; ++h) {
    int c  = tid + h * 256;            // chunk 0..511
    int r  = c >> 2;                   // tile row 0..127
    int ko = (c & 3) * 8;              // k offset (elements)
    gx[h] = xb + (size_t)(mb + r) * K_DIM + ko;
    gl[h] = lb + (size_t)(mb + r) * K_DIM + ko;
    ldx0[h] = (uint32_t)(uintptr_t)&sX[0][r * LDSROW + ko];
    ldl0[h] = (uint32_t)(uintptr_t)&sL[0][r * LDSROW + ko];
  }

  // ---- TDM descriptor group 1 (loop-invariant): 2-byte elems, LDS padding
  //      16B every 64B row, tile 32(k) x 64(n), row stride K_DIM.
  u32x8 g1;
  g1[0] = (1u << 16)                   // data_size = 2 bytes
        | (1u << 20)                   // pad_enable
        | (3u << 22)                   // pad_interval: 16 DWORDs (64 B)
        | (3u << 25);                  // pad_amount:   4 DWORDs (16 B)
  g1[1] = (uint32_t)(K_DIM & 0xFFFFu) << 16;                       // dim0 lo
  g1[2] = (uint32_t)(K_DIM >> 16) | ((uint32_t)(N_DIM & 0xFFFFu) << 16);
  g1[3] = (uint32_t)(N_DIM >> 16) | (32u << 16);                   // tile_dim0
  g1[4] = 64u;                         // tile_dim1 (no tile_dim2)
  g1[5] = (uint32_t)K_DIM;             // tensor_dim0_stride (elements)
  g1[6] = 0u;
  g1[7] = 0u;

  const uint32_t ldsw0 = (uint32_t)(uintptr_t)&sW[0][0];
  const uint32_t ldsg0 = (uint32_t)(uintptr_t)&sG[0][0];
  const uint16_t* wbase = wt + (size_t)nb * K_DIM;
  const uint16_t* gbase = gt + (size_t)nb * K_DIM;

  // Stage slab at element offset KK into buffer BUFI (literal 0/1).
#define STAGE_ALL(BUFI, KK)                                                    \
  do {                                                                         \
    _Pragma("unroll")                                                          \
    for (int h = 0; h < 2; ++h) {                                              \
      uint64_t ax = (uint64_t)(uintptr_t)(gx[h] + (KK));                       \
      uint64_t al = (uint64_t)(uintptr_t)(gl[h] + (KK));                       \
      uint32_t dx = ldx0[h] + (BUFI) * (128 * LDSROW * 2);                     \
      uint32_t dl = ldl0[h] + (BUFI) * (128 * LDSROW * 2);                     \
      asm volatile("global_load_async_to_lds_b128 %0, %1, off"                 \
                   :: "v"(dx), "v"(ax) : "memory");                            \
      asm volatile("global_load_async_to_lds_b128 %0, %1, off"                 \
                   :: "v"(dl), "v"(al) : "memory");                            \
    }                                                                          \
    if (wave == 0) {                                                           \
      u32x4 gw0 = tdm_g0(ldsw0 + (BUFI) * (64 * LDSROW * 2),                   \
                         (uint64_t)(uintptr_t)(wbase + (KK)));                 \
      u32x4 gg0 = tdm_g0(ldsg0 + (BUFI) * (64 * LDSROW * 2),                   \
                         (uint64_t)(uintptr_t)(gbase + (KK)));                 \
      asm volatile("tensor_load_to_lds %0, %1"                                 \
                   :: "s"(gw0), "s"(g1) : "memory");                           \
      asm volatile("tensor_load_to_lds %0, %1"                                 \
                   :: "s"(gg0), "s"(g1) : "memory");                           \
    }                                                                          \
  } while (0)

#define WAIT_SLAB(N_ASYNC, N_TENSOR)                                           \
  do {                                                                         \
    asm volatile("s_wait_asynccnt " #N_ASYNC ::: "memory");                    \
    if (wave == 0) __builtin_amdgcn_s_wait_tensorcnt(N_TENSOR);                \
  } while (0)

  v8f acc_a[2][2] = {};
  v8f acc_m[2][2] = {};
  v8f acc_g[2][2] = {};

  // per-wave LDS fragment base pointers (buf0; buf1 is a constant offset)
  const uint16_t* pX = &sX[0][(wm * 32 + lr) * LDSROW + lh * 8];
  const uint16_t* pL = &sL[0][(wm * 32 + lr) * LDSROW + lh * 8];
  const uint16_t* pW = &sW[0][(wn * 32 + lr) * LDSROW + lh * 8];
  const uint16_t* pG = &sG[0][(wn * 32 + lr) * LDSROW + lh * 8];

#define COMPUTE(BUFI)                                                          \
  do {                                                                         \
    v16bf fx[2], fl[2], fw[2], fg[2];                                          \
    _Pragma("unroll")                                                          \
    for (int t = 0; t < 2; ++t) {                                              \
      fx[t] = load_frag(pX + (BUFI) * 128 * LDSROW + t * 16 * LDSROW);         \
      fl[t] = load_frag(pL + (BUFI) * 128 * LDSROW + t * 16 * LDSROW);         \
    }                                                                          \
    _Pragma("unroll")                                                          \
    for (int u = 0; u < 2; ++u) {                                              \
      fw[u] = load_frag(pW + (BUFI) * 64 * LDSROW + u * 16 * LDSROW);          \
      fg[u] = load_frag(pG + (BUFI) * 64 * LDSROW + u * 16 * LDSROW);          \
    }                                                                          \
    _Pragma("unroll")                                                          \
    for (int t = 0; t < 2; ++t) {                                              \
      _Pragma("unroll")                                                        \
      for (int u = 0; u < 2; ++u) {                                            \
        acc_a[t][u] = __builtin_amdgcn_wmma_f32_16x16x32_bf16(                 \
            false, fx[t], false, fw[u], (short)0, acc_a[t][u], false, false);  \
        acc_m[t][u] = __builtin_amdgcn_wmma_f32_16x16x32_bf16(                 \
            false, fl[t], false, fw[u], (short)0, acc_m[t][u], false, false);  \
        acc_g[t][u] = __builtin_amdgcn_wmma_f32_16x16x32_bf16(                 \
            false, fx[t], false, fg[u], (short)0, acc_g[t][u], false, false);  \
      }                                                                        \
    }                                                                          \
  } while (0)

  const int NK = K_DIM / 32;           // 64 slabs (even)
  STAGE_ALL(0, 0);                     // prologue: slab 0 -> buf0

  for (int ks = 0; ks < NK; ks += 2) {
    const int k = ks * 32;
    // phase A: prefetch slab ks+1 -> buf1; compute slab ks from buf0
    STAGE_ALL(1, k + 32);
    WAIT_SLAB(0x4, 2);                 // retire oldest slab only
    __syncthreads();
    COMPUTE(0);
    __syncthreads();
    // phase B: prefetch slab ks+2 -> buf0; compute slab ks+1 from buf1
    if (ks + 2 < NK) {
      STAGE_ALL(0, k + 64);
      WAIT_SLAB(0x4, 2);
    } else {
      WAIT_SLAB(0x0, 0);
    }
    __syncthreads();
    COMPUTE(1);
    __syncthreads();
  }
#undef STAGE_ALL
#undef WAIT_SLAB
#undef COMPUTE

  // epilogue: out = sigmoid(g)*a + (1-sigmoid(g))*exp(m) = g*(a-m)+m
  // C/D layout: VGPR r -> row (r + lh*8), col = lr within the 16x16 tile.
#pragma unroll
  for (int t = 0; t < 2; ++t) {
#pragma unroll
    for (int u = 0; u < 2; ++u) {
      float* op = out + (size_t)(m0 + t * 16 + lh * 8) * N_DIM
                      + (nb + wn * 32 + u * 16 + lr);
#pragma unroll
      for (int r = 0; r < 8; ++r) {
        float g = fast_sigmoid(acc_g[t][u][r]);
        float m = fast_exp(acc_m[t][u][r]);
        op[(size_t)r * N_DIM] = g * (acc_a[t][u][r] - m) + m;
      }
    }
  }
}

// ---------- launch ----------

extern "C" void kernel_launch(void* const* d_in, const int* in_sizes, int n_in,
                              void* d_out, int out_size, void* d_ws, size_t ws_size,
                              hipStream_t stream) {
  const float* x  = (const float*)d_in[0];   // [8192][2048]
  const float* Wh = (const float*)d_in[1];   // [2048][2048]
  const float* Mh = (const float*)d_in[2];   // [2048][2048]
  const float* G  = (const float*)d_in[3];   // [2048][2048]
  float* out = (float*)d_out;

  // workspace carve-up (bf16 staging): 2*B*K + 2*K*N ushorts = ~84 MB
  uint16_t* xb = (uint16_t*)d_ws;
  uint16_t* lb = xb + (size_t)B_DIM * K_DIM;
  uint16_t* wt = lb + (size_t)B_DIM * K_DIM;
  uint16_t* gt = wt + (size_t)K_DIM * N_DIM;

  nalu_prep_x<<<(B_DIM * K_DIM / 4) / 256, 256, 0, stream>>>(x, xb, lb);
  nalu_prep_w<<<dim3(N_DIM / 32, K_DIM / 32), dim3(32, 8), 0, stream>>>(Wh, Mh, G, wt, gt);
  nalu_gemm<<<dim3(N_DIM / 64, B_DIM / 128), 256, 0, stream>>>(xb, lb, wt, gt, out);
}
